// Qwen2MLP_44976897524099
// MI455X (gfx1250) — compile-verified, matrix operand order checked by reading, not run on hardware
//
#include <hip/hip_runtime.h>

// ---------------------------------------------------------------------------
// Qwen2 MLP, AWQ int4 weights, fp16 activations, WMMA f16 (gfx1250).
//   k1: gate_up GEMM + SiLU*up fusion  -> intermediate (fp16, in d_ws)
//   k2: down GEMM                      -> d_out (fp16)
// Packed-fp16 "magic number" dequant, per-group hoisted zeros/scales,
// async global->LDS staging of the activation tile when available.
// ---------------------------------------------------------------------------

typedef _Float16 half_t;
typedef __attribute__((ext_vector_type(2)))  _Float16 h2;
typedef __attribute__((ext_vector_type(8)))  _Float16 v8h;
typedef __attribute__((ext_vector_type(16))) _Float16 v16h;
typedef __attribute__((ext_vector_type(8)))  float    v8f;
typedef __attribute__((ext_vector_type(4)))  int      v4i;

#define M_TOT   4096
#define GU_K    3584
#define GU_NC   4736      // qweight_gu int32 columns (37888/8)
#define GU_N    37888
#define INTER_N 18944
#define DN_K    18944
#define DN_NC   448       // qweight_down int32 columns (3584/8)
#define DN_N    3584
#define KT      64        // K per LDS stage (2 WMMA k-steps)
#define BM      128       // block tile M
#define BN      64        // block tile N

#if defined(__has_builtin)
#if __has_builtin(__builtin_amdgcn_global_load_async_to_lds_b128) && \
    __has_builtin(__builtin_amdgcn_s_wait_asynccnt)
#define USE_ASYNC_LDS 1
#endif
#endif

// 16-byte global -> LDS copy; async DMA (ASYNCcnt) when the builtin exists.
__device__ __forceinline__ void cp16_g2l(const half_t* gsrc, half_t* ldst) {
#if defined(USE_ASYNC_LDS)
    __builtin_amdgcn_global_load_async_to_lds_b128(
        (__attribute__((address_space(1))) v4i*)(gsrc),
        (__attribute__((address_space(3))) v4i*)(ldst), 0, 0);
#else
    *(v8h*)(ldst) = *(const v8h*)(gsrc);
#endif
}

__device__ __forceinline__ void cp_wait() {
#if defined(USE_ASYNC_LDS)
    __builtin_amdgcn_s_wait_asynccnt(0);
#endif
}

// fp16 magic: 0x6400 | n  ==  (half)(1024 + n)  for n in [0,16)
__device__ __forceinline__ h2 magic_pair(unsigned lo, unsigned hi) {
    union { unsigned u; h2 h; } c;
    c.u = 0x64006400u | lo | (hi << 16);
    return c.h;
}

// One 16x32 WMMA fragment from an LDS tile with 64-half row stride.
// Lane l: row = l&15, h = l>>4; halves 0..7 = K [8h,8h+8),
// halves 8..15 = K [16+8h, 16+8h+8)  (two ds_load_b128).
__device__ __forceinline__ v16h lds_frag(const half_t* base, int row, int kb, int h) {
    union { v16h v; v8h p[2]; } u;
    const int off = row * KT + kb + 8 * h;
    u.p[0] = *(const v8h*)(base + off);
    u.p[1] = *(const v8h*)(base + off + 16);
    return u.v;
}

#define WMMA(acc, a, b) \
    acc = __builtin_amdgcn_wmma_f32_16x16x32_f16(false, a, false, b, (short)0, acc, false, false)

// ---------------------------------------------------------------------------
// Kernel 1: gate_up = x @ dequant(W_gu); inter = silu(gate)*up
// grid: (INTER_N/BN, M_TOT/BM), block: 256 (8 waves = 4M x 2N)
// ---------------------------------------------------------------------------
__global__ __launch_bounds__(256)
void mlp_gateup_silu(const half_t* __restrict__ x,
                     const int*    __restrict__ qw,
                     const int*    __restrict__ qz,
                     const half_t* __restrict__ sc,
                     half_t*       __restrict__ inter) {
    __shared__ half_t As[BM * KT];   // 16 KB
    __shared__ half_t Bg[BN * KT];   //  8 KB
    __shared__ half_t Bu[BN * KT];   //  8 KB

    const int tid  = threadIdx.x;
    const int lane = tid & 31;
    const int wave = tid >> 5;
    const int wm   = wave & 3;      // 4 M-subtiles of 32 rows
    const int wn   = wave >> 2;     // 2 N-subtiles of 32 cols
    const int m0   = blockIdx.y * BM;
    const int n0   = blockIdx.x * BN;     // gate column base

    // A stager: thread -> row m, 32 halves (4x b128)
    const int a_m = tid >> 1;
    const int a_k = (tid & 1) * 32;
    // B dequant: thread -> k-pair kp, packed-int col nc
    const int kp  = tid >> 3;             // 0..31 -> k rows 2kp,2kp+1
    const int nc  = tid & 7;              // int32 within 64-col tile
    const int nc8 = nc * 8;
    const int kk2 = kp * 2;
    const int ncg = (n0 >> 3) + nc;                 // gate packed col
    const int ncu = ((n0 + INTER_N) >> 3) + nc;     // up packed col

    const int frow = lane & 15;
    const int fh   = lane >> 4;

    v8f accg[2][2] = {{{}, {}}, {{}, {}}};
    v8f accu[2][2] = {{{}, {}}, {{}, {}}};

    for (int g = 0; g < GU_K / 128; ++g) {
        // ---- per-group zero/scale prep (hoisted over 2 stages) ----
        const unsigned zg = (unsigned)qz[(size_t)g * GU_NC + ncg];
        const unsigned zu = (unsigned)qz[(size_t)g * GU_NC + ncu];
        const v8h svg = *(const v8h*)(sc + (size_t)g * GU_N + n0 + nc8);
        const v8h svu = *(const v8h*)(sc + (size_t)g * GU_N + n0 + INTER_N + nc8);
        h2 zpg[8], zpu[8], sg2[8], su2[8];
#pragma unroll
        for (int j = 0; j < 8; ++j) {
            const unsigned zgn = (zg >> (4 * j)) & 15u;
            const unsigned zun = (zu >> (4 * j)) & 15u;
            zpg[j] = magic_pair(zgn, zgn);
            zpu[j] = magic_pair(zun, zun);
            sg2[j] = (h2){svg[j], svg[j]};
            su2[j] = (h2){svu[j], svu[j]};
        }

        for (int s = 0; s < 2; ++s) {
            const int k0 = g * 128 + s * KT;

            // ---- async-stage A tile (activations, pure copy) ----
            const half_t* gA = x + (size_t)(m0 + a_m) * GU_K + k0 + a_k;
            half_t*       lA = As + a_m * KT + a_k;
#pragma unroll
            for (int q = 0; q < 4; ++q) cp16_g2l(gA + q * 8, lA + q * 8);

            // ---- dequant gate+up B tiles into LDS [n][k] ----
            const size_t qrow = (size_t)(k0 + kk2) * GU_NC;
            const unsigned qg0 = (unsigned)qw[qrow + ncg];
            const unsigned qg1 = (unsigned)qw[qrow + GU_NC + ncg];
            const unsigned qu0 = (unsigned)qw[qrow + ncu];
            const unsigned qu1 = (unsigned)qw[qrow + GU_NC + ncu];
#pragma unroll
            for (int j = 0; j < 8; ++j) {
                const unsigned sh = 4 * j;
                h2 rg = (magic_pair((qg0 >> sh) & 15u, (qg1 >> sh) & 15u) - zpg[j]) * sg2[j];
                h2 ru = (magic_pair((qu0 >> sh) & 15u, (qu1 >> sh) & 15u) - zpu[j]) * su2[j];
                *(h2*)(&Bg[(nc8 + j) * KT + kk2]) = rg;
                *(h2*)(&Bu[(nc8 + j) * KT + kk2]) = ru;
            }

            if (k0 + KT < GU_K) {   // pull next stage's packed rows toward L2
                __builtin_prefetch((const void*)(qw + qrow + (size_t)KT * GU_NC + ncg), 0, 1);
                __builtin_prefetch((const void*)(qw + qrow + (size_t)KT * GU_NC + ncu), 0, 1);
            }

            cp_wait();
            __syncthreads();

            // ---- 16 WMMAs per wave per stage ----
#pragma unroll
            for (int kb = 0; kb < KT; kb += 32) {
                const v16h a0  = lds_frag(As, wm * 32 + frow,      kb, fh);
                const v16h a1  = lds_frag(As, wm * 32 + 16 + frow, kb, fh);
                const v16h bg0 = lds_frag(Bg, wn * 32 + frow,      kb, fh);
                const v16h bg1 = lds_frag(Bg, wn * 32 + 16 + frow, kb, fh);
                const v16h bu0 = lds_frag(Bu, wn * 32 + frow,      kb, fh);
                const v16h bu1 = lds_frag(Bu, wn * 32 + 16 + frow, kb, fh);
                WMMA(accg[0][0], a0, bg0); WMMA(accg[0][1], a0, bg1);
                WMMA(accg[1][0], a1, bg0); WMMA(accg[1][1], a1, bg1);
                WMMA(accu[0][0], a0, bu0); WMMA(accu[0][1], a0, bu1);
                WMMA(accu[1][0], a1, bu0); WMMA(accu[1][1], a1, bu1);
            }

            __syncthreads();
        }
    }

    // ---- epilogue: silu(gate)*up -> intermediate (fp16) ----
    const int cn = lane & 15;
    const int ch = lane >> 4;
#pragma unroll
    for (int mi = 0; mi < 2; ++mi) {
#pragma unroll
        for (int ni = 0; ni < 2; ++ni) {
#pragma unroll
            for (int r = 0; r < 8; ++r) {
                const int m = m0 + wm * 32 + mi * 16 + r + 8 * ch;
                const int n = n0 + wn * 32 + ni * 16 + cn;
                const float gg = accg[mi][ni][r];
                const float uu = accu[mi][ni][r];
                inter[(size_t)m * INTER_N + n] =
                    (half_t)(gg / (1.0f + __expf(-gg)) * uu);
            }
        }
    }
}

// ---------------------------------------------------------------------------
// Kernel 2: out = inter @ dequant(W_down)
// grid: (DN_N/BN, M_TOT/BM), block: 256
// ---------------------------------------------------------------------------
__global__ __launch_bounds__(256)
void mlp_down(const half_t* __restrict__ inter,
              const int*    __restrict__ qw,
              const int*    __restrict__ qz,
              const half_t* __restrict__ sc,
              half_t*       __restrict__ out) {
    __shared__ half_t As[BM * KT];   // 16 KB
    __shared__ half_t Bs[BN * KT];   //  8 KB

    const int tid  = threadIdx.x;
    const int lane = tid & 31;
    const int wave = tid >> 5;
    const int wm   = wave & 3;
    const int wn   = wave >> 2;
    const int m0   = blockIdx.y * BM;
    const int n0   = blockIdx.x * BN;

    const int a_m = tid >> 1;
    const int a_k = (tid & 1) * 32;
    const int kp  = tid >> 3;
    const int nc  = tid & 7;
    const int nc8 = nc * 8;
    const int kk2 = kp * 2;
    const int ncd = (n0 >> 3) + nc;

    const int frow = lane & 15;
    const int fh   = lane >> 4;

    v8f acc[2][2] = {{{}, {}}, {{}, {}}};

    for (int g = 0; g < DN_K / 128; ++g) {
        const unsigned zw = (unsigned)qz[(size_t)g * DN_NC + ncd];
        const v8h sv = *(const v8h*)(sc + (size_t)g * DN_N + n0 + nc8);
        h2 zp[8], s2[8];
#pragma unroll
        for (int j = 0; j < 8; ++j) {
            const unsigned zn = (zw >> (4 * j)) & 15u;
            zp[j] = magic_pair(zn, zn);
            s2[j] = (h2){sv[j], sv[j]};
        }

        for (int s = 0; s < 2; ++s) {
            const int k0 = g * 128 + s * KT;

            const half_t* gA = inter + (size_t)(m0 + a_m) * DN_K + k0 + a_k;
            half_t*       lA = As + a_m * KT + a_k;
#pragma unroll
            for (int q = 0; q < 4; ++q) cp16_g2l(gA + q * 8, lA + q * 8);

            const size_t qrow = (size_t)(k0 + kk2) * DN_NC;
            const unsigned q0 = (unsigned)qw[qrow + ncd];
            const unsigned q1 = (unsigned)qw[qrow + DN_NC + ncd];
#pragma unroll
            for (int j = 0; j < 8; ++j) {
                const unsigned sh = 4 * j;
                h2 r = (magic_pair((q0 >> sh) & 15u, (q1 >> sh) & 15u) - zp[j]) * s2[j];
                *(h2*)(&Bs[(nc8 + j) * KT + kk2]) = r;
            }

            if (k0 + KT < DN_K) {
                __builtin_prefetch((const void*)(qw + qrow + (size_t)KT * DN_NC + ncd), 0, 1);
            }

            cp_wait();
            __syncthreads();

#pragma unroll
            for (int kb = 0; kb < KT; kb += 32) {
                const v16h a0 = lds_frag(As, wm * 32 + frow,      kb, fh);
                const v16h a1 = lds_frag(As, wm * 32 + 16 + frow, kb, fh);
                const v16h b0 = lds_frag(Bs, wn * 32 + frow,      kb, fh);
                const v16h b1 = lds_frag(Bs, wn * 32 + 16 + frow, kb, fh);
                WMMA(acc[0][0], a0, b0); WMMA(acc[0][1], a0, b1);
                WMMA(acc[1][0], a1, b0); WMMA(acc[1][1], a1, b1);
            }

            __syncthreads();
        }
    }

    const int cn = lane & 15;
    const int ch = lane >> 4;
#pragma unroll
    for (int mi = 0; mi < 2; ++mi) {
#pragma unroll
        for (int ni = 0; ni < 2; ++ni) {
#pragma unroll
            for (int r = 0; r < 8; ++r) {
                const int m = m0 + wm * 32 + mi * 16 + r + 8 * ch;
                out[(size_t)m * DN_N + n0 + wn * 32 + ni * 16 + cn] =
                    (half_t)acc[mi][ni][r];
            }
        }
    }
}

// ---------------------------------------------------------------------------
extern "C" void kernel_launch(void* const* d_in, const int* in_sizes, int n_in,
                              void* d_out, int out_size, void* d_ws, size_t ws_size,
                              hipStream_t stream) {
    (void)in_sizes; (void)n_in; (void)out_size; (void)ws_size;

    const half_t* x     = (const half_t*)d_in[0];
    const int*    qw_gu = (const int*)   d_in[1];
    const int*    qz_gu = (const int*)   d_in[2];
    const half_t* sc_gu = (const half_t*)d_in[3];
    const int*    qw_dn = (const int*)   d_in[4];
    const int*    qz_dn = (const int*)   d_in[5];
    const half_t* sc_dn = (const half_t*)d_in[6];

    half_t* out   = (half_t*)d_out;
    half_t* inter = (half_t*)d_ws;      // 4096*18944 fp16 = 155 MB scratch

    dim3 blk(256);
    dim3 g1(INTER_N / BN, M_TOT / BM);  // 296 x 32
    hipLaunchKernelGGL(mlp_gateup_silu, g1, blk, 0, stream,
                       x, qw_gu, qz_gu, sc_gu, inter);

    dim3 g2(DN_N / BN, M_TOT / BM);     // 56 x 32
    hipLaunchKernelGGL(mlp_down, g2, blk, 0, stream,
                       inter, qw_dn, qz_dn, sc_dn, out);
}
